// CNODExtmod_70781061038895
// MI455X (gfx1250) — matrix operand order repaired and checked by reading, not compiled, not run on hardware
//
#include <hip/hip_runtime.h>

typedef __attribute__((ext_vector_type(16))) _Float16 v16h;
typedef __attribute__((ext_vector_type(8)))  float    v8f;

#define NC     32
#define HID    128
#define IN1    33          // Nc + 1
#define TSTEPS 64
#define NSUB   4
#define BATCHN 4096
#define BM     16          // batch rows per workgroup
#define DTC    0.01f

// ---------------- LDS layout (byte offsets into dynamic smem) ----------------
// All f16 tile rows are 16B-aligned so fragments load as ds_load_b128 pairs.
#define OFF_A    0          // f32 [32][32]                        4096 B
#define OFF_BV   4096       // f32 [32]                             128 B
#define OFF_B1   4224       // f32 [128]                            512 B
#define OFF_B2   4736       // f32 [128]                            512 B
#define OFF_B3   5248       // f32 [48] (zeros past 33)             192 B
#define OFF_W1   5440       // f16 [128][PW1=40]                  10240 B
#define PW1      40
#define OFF_W2   15680      // f16 [128][128]                     32768 B
#define PW2      128
#define OFF_W3   48448      // f16 [48][128] (rows 33..47 zero)   12288 B
#define PW3      128
#define OFF_X    60736      // f16 [16][PX=40]  activations cn1    1280 B
#define PX       40
#define OFF_H1   62016      // f16 [16][PH=136] hidden1            4352 B
#define PH       136
#define OFF_H2   66368      // f16 [16][136]    hidden2            4352 B
#define OFF_C0   70720      // f32 [16][32]     cn0 state          2048 B
#define OFF_C1   72768      // f32 [16][33]     cn1 state          2112 B
#define OFF_D0   74880      // f32 [16][32]     d cn0              2048 B
#define OFF_D1   76928      // f32 [16][PD1=48] d cn1              3072 B
#define PD1      48
#define OFF_YT   80000      // f32 [16][64]     Y tile             4096 B
#define OFF_MT   84096      // f32 [16][64]     mask tile          4096 B
#define LDS_BYTES 88192

// ---- WMMA A-matrix fragment (16x32 f16) from weight tile [N][K] row-major ----
// Per lane: halves (k0 + hi*8 .. +7) and (k0 + 16 + hi*8 .. +7)  -> 2x b128.
__device__ inline v16h load_w_frag(const _Float16* Wt, int pitch, int k0, int lane) {
    const int r  = lane & 15;
    const int hi = lane >> 4;
    union { v16h v; int4 q[2]; } f;
    const char* base = (const char*)(Wt + r * pitch + k0 + hi * 8);
    f.q[0] = *(const int4*)base;          // VGPR 0..3  (K k0+hi*8 .. +7)
    f.q[1] = *(const int4*)(base + 32);   // VGPR 4..7  (K k0+16+hi*8 .. +7)
    return f.v;
}

// ---- WMMA B-matrix fragment (32x16 f16) = Act^T from Act [M][K] row-major ----
// Per lane (col m = lane&15): halves (k0 + hi*8 .. +7) and (k0 + 16 + hi*8 .. +7) -> 2x b128.
__device__ inline v16h load_act_frag(const _Float16* Act, int pitch, int k0, int lane) {
    const int m  = lane & 15;
    const int hi = lane >> 4;
    union { v16h v; int4 q[2]; } f;
    const char* base = (const char*)(Act + m * pitch + k0 + hi * 8);
    f.q[0] = *(const int4*)base;
    f.q[1] = *(const int4*)(base + 32);
    return f.v;
}

__global__ __launch_bounds__(256)
void cnode_persistent_kernel(const float* __restrict__ times,
                             const float* __restrict__ Y,
                             const float* __restrict__ mask,
                             const float* __restrict__ A,
                             const float* __restrict__ Bvec,
                             const float* __restrict__ W1, const float* __restrict__ b1,
                             const float* __restrict__ W2, const float* __restrict__ b2,
                             const float* __restrict__ W3, const float* __restrict__ b3,
                             float* __restrict__ out)
{
    extern __shared__ char smem[];
    float*    sA  = (float*)   (smem + OFF_A);
    float*    sBv = (float*)   (smem + OFF_BV);
    float*    sb1 = (float*)   (smem + OFF_B1);
    float*    sb2 = (float*)   (smem + OFF_B2);
    float*    sb3 = (float*)   (smem + OFF_B3);
    _Float16* sW1 = (_Float16*)(smem + OFF_W1);
    _Float16* sW2 = (_Float16*)(smem + OFF_W2);
    _Float16* sW3 = (_Float16*)(smem + OFF_W3);
    _Float16* sX  = (_Float16*)(smem + OFF_X);
    _Float16* sH1 = (_Float16*)(smem + OFF_H1);
    _Float16* sH2 = (_Float16*)(smem + OFF_H2);
    float*    sc0 = (float*)   (smem + OFF_C0);
    float*    sc1 = (float*)   (smem + OFF_C1);
    float*    sd0 = (float*)   (smem + OFF_D0);
    float*    sd1 = (float*)   (smem + OFF_D1);
    float*    sYt = (float*)   (smem + OFF_YT);
    float*    sMt = (float*)   (smem + OFF_MT);

    const int tid  = threadIdx.x;
    const int lane = tid & 31;
    const int w    = tid >> 5;              // wave id 0..7; owns output-channel tile [w*16, w*16+16)
    const int hi   = lane >> 4;
    const int ml   = lane & 15;             // batch row this lane's D column maps to
    const int nb   = w * 16 + 8 * hi;       // base output channel of this lane's 8 D rows
    const int row0 = blockIdx.x * BM;       // first batch row of this block

    // ---------------- one-time staging into LDS ----------------
    for (int i = tid; i < HID * IN1; i += 256) sW1[(i / IN1) * PW1 + (i % IN1)] = (_Float16)W1[i];
    for (int i = tid; i < HID * HID; i += 256) sW2[i] = (_Float16)W2[i];
    for (int i = tid; i < 48 * HID;  i += 256) sW3[i] = (i < IN1 * HID) ? (_Float16)W3[i] : (_Float16)0.0f;
    for (int i = tid; i < NC * NC;   i += 256) sA[i] = A[i];
    if (tid < NC)  sBv[tid] = Bvec[tid];
    if (tid < HID) { sb1[tid] = b1[tid]; sb2[tid] = b2[tid]; }
    if (tid < 48)  sb3[tid] = (tid < IN1) ? b3[tid] : 0.0f;
    for (int i = tid; i < BM * NC;  i += 256) sc0[i] = 0.0f;
    for (int i = tid; i < BM * IN1; i += 256) {
        const int m = i / IN1, ii = i - m * IN1;
        sc1[i] = 0.0f;
        sX[m * PX + ii] = (_Float16)0.0f;
    }
    for (int i = tid; i < BM * TSTEPS; i += 256) {          // i = m*64 + t, coalesced
        sYt[i] = Y[row0 * TSTEPS + i];
        sMt[i] = mask[row0 * TSTEPS + i];
    }
    __syncthreads();

    // ---------------- hoist loop-invariant weight fragments into VGPRs ----------------
    const v16h vzero = {};
    v16h w1a = load_w_frag(sW1 + (w * 16) * PW1, PW1, 0, lane);
    v16h w2a[4], w3a[4];
#pragma unroll
    for (int kc = 0; kc < 4; ++kc)
        w2a[kc] = load_w_frag(sW2 + (w * 16) * PW2, PW2, kc * 32, lane);
    if (w < 3) {
#pragma unroll
        for (int kc = 0; kc < 4; ++kc)
            w3a[kc] = load_w_frag(sW3 + (w * 16) * PW3, PW3, kc * 32, lane);
    } else {
#pragma unroll
        for (int kc = 0; kc < 4; ++kc) w3a[kc] = vzero;
    }
    float w1l[8], bb1[8], bb2[8], bb3[8];
#pragma unroll
    for (int v = 0; v < 8; ++v) {
        const int n = nb + v;
        w1l[v] = (float)sW1[n * PW1 + 32];   // K=32 column of W1 (rank-1 term)
        bb1[v] = sb1[n];
        bb2[v] = sb2[n];
    }
    if (w < 3) {
#pragma unroll
        for (int v = 0; v < 8; ++v) bb3[v] = sb3[nb + v];
    } else {
#pragma unroll
        for (int v = 0; v < 8; ++v) bb3[v] = 0.0f;
    }

#pragma unroll 1
    for (int t = 0; t < TSTEPS; ++t) {
#pragma unroll 1
        for (int s = 0; s < NSUB; ++s) {
            // ---- phase A: layer 1: D^T(n,m) = W1 x X^T ; WMMA K=32 + rank-1 ----
            {
                v16h bx = load_act_frag(sX, PX, 0, lane);
                const float x32 = (float)sX[ml * PX + 32];
                v8f c = {};
                c = __builtin_amdgcn_wmma_f32_16x16x32_f16(false, w1a, false, bx, (short)0, c, false, false);
                union { _Float16 h[8]; int4 q; } pk;
#pragma unroll
                for (int v = 0; v < 8; ++v) {
                    float r = c[v] + w1l[v] * x32 + bb1[v];
                    r = r > 0.0f ? r : 0.0f;
                    pk.h[v] = (_Float16)r;
                }
                *((int4*)(sH1 + ml * PH + nb)) = pk.q;   // 8 consecutive halves: 1x b128
            }
            __syncthreads();

            // ---- phase B: layer 2, all 4 B-fragments prefetched then WMMA chain ----
            {
                v16h bx0 = load_act_frag(sH1, PH,  0, lane);
                v16h bx1 = load_act_frag(sH1, PH, 32, lane);
                v16h bx2 = load_act_frag(sH1, PH, 64, lane);
                v16h bx3 = load_act_frag(sH1, PH, 96, lane);
                v8f c = {};
                c = __builtin_amdgcn_wmma_f32_16x16x32_f16(false, w2a[0], false, bx0, (short)0, c, false, false);
                c = __builtin_amdgcn_wmma_f32_16x16x32_f16(false, w2a[1], false, bx1, (short)0, c, false, false);
                c = __builtin_amdgcn_wmma_f32_16x16x32_f16(false, w2a[2], false, bx2, (short)0, c, false, false);
                c = __builtin_amdgcn_wmma_f32_16x16x32_f16(false, w2a[3], false, bx3, (short)0, c, false, false);
                union { _Float16 h[8]; int4 q; } pk;
#pragma unroll
                for (int v = 0; v < 8; ++v) {
                    float r = c[v] + bb2[v];
                    r = r > 0.0f ? r : 0.0f;
                    pk.h[v] = (_Float16)r;
                }
                *((int4*)(sH2 + ml * PH + nb)) = pk.q;
            }
            __syncthreads();

            // ---- phase C: layer 3 on waves 0..2  ||  f32 linear dynamics on waves 3..7 ----
            if (w < 3) {
                v16h bx0 = load_act_frag(sH2, PH,  0, lane);
                v16h bx1 = load_act_frag(sH2, PH, 32, lane);
                v16h bx2 = load_act_frag(sH2, PH, 64, lane);
                v16h bx3 = load_act_frag(sH2, PH, 96, lane);
                v8f c = {};
                c = __builtin_amdgcn_wmma_f32_16x16x32_f16(false, w3a[0], false, bx0, (short)0, c, false, false);
                c = __builtin_amdgcn_wmma_f32_16x16x32_f16(false, w3a[1], false, bx1, (short)0, c, false, false);
                c = __builtin_amdgcn_wmma_f32_16x16x32_f16(false, w3a[2], false, bx2, (short)0, c, false, false);
                c = __builtin_amdgcn_wmma_f32_16x16x32_f16(false, w3a[3], false, bx3, (short)0, c, false, false);
                float* dst = sd1 + ml * PD1 + nb;
                *((float4*)dst)       = make_float4(c[0] + bb3[0], c[1] + bb3[1], c[2] + bb3[2], c[3] + bb3[3]);
                *((float4*)(dst + 4)) = make_float4(c[4] + bb3[4], c[5] + bb3[5], c[6] + bb3[6], c[7] + bb3[7]);
            } else {
                // d_cn0 = cn0 @ A^T + Bvec * cn1[:,0]  (reads state, stable until phase D)
                const int tid2 = tid - 96;                  // 0..159 across waves 3..7
                for (int e = tid2; e < BM * NC; e += 160) {
                    const int m = e >> 5, j = e & 31;
                    float acc = sBv[j] * sc1[m * IN1];
                    const float* arow = sA + j * NC;
                    const float* crow = sc0 + m * NC;
#pragma unroll 8
                    for (int k = 0; k < NC; ++k) acc += crow[k] * arow[k];
                    sd0[e] = acc;
                }
            }
            __syncthreads();

            // ---- phase D: Euler update; also refresh f16 copy of cn1 ----
            for (int e = tid; e < BM * NC; e += 256) sc0[e] += DTC * sd0[e];
            for (int e = tid; e < BM * IN1; e += 256) {
                const int m = e / IN1, i = e - m * IN1;
                const float nv = sc1[e] + DTC * sd1[m * PD1 + i];
                sc1[e] = nv;
                sX[m * PX + i] = (_Float16)nv;
            }
            __syncthreads();
        } // substeps

        // ---- observation: emit y_pred (old cn1[:,0]) then masked state update ----
        for (int e = tid; e < BM * IN1; e += 256) {
            const int m = e / IN1, i = e - m * IN1;
            const float mv  = sMt[m * TSTEPS + t];
            const float cur = sc1[e];
            if (i == 0) {                       // only this thread touches cn1[m,0]
                const int gi = (row0 + m) * TSTEPS + t;
                out[gi] = cur;
                out[BATCHN * TSTEPS + gi] = cur;
            }
            const float upd = (i == 0) ? sYt[m * TSTEPS + t] : sc0[m * NC + (i - 1)];
            const float nv  = mv * upd + (1.0f - mv) * cur;
            sc1[e] = nv;
            sX[m * PX + i] = (_Float16)nv;
        }
        __syncthreads();
    } // t

    // ---- final outputs: times passthrough and hT[:, :Nc] ----
    for (int e = tid; e < BM * NC; e += 256) {
        const int m = e >> 5, j = e & 31;
        out[BATCHN * TSTEPS * 2 + TSTEPS + (row0 + m) * NC + j] = sc0[e];
    }
    if (blockIdx.x == 0 && tid < TSTEPS)
        out[BATCHN * TSTEPS * 2 + tid] = times[tid];
}

extern "C" void kernel_launch(void* const* d_in, const int* in_sizes, int n_in,
                              void* d_out, int out_size, void* d_ws, size_t ws_size,
                              hipStream_t stream) {
    const float* times = (const float*)d_in[0];
    const float* Y     = (const float*)d_in[1];
    const float* mask  = (const float*)d_in[2];
    const float* A     = (const float*)d_in[3];
    const float* Bvec  = (const float*)d_in[4];
    const float* W1    = (const float*)d_in[5];
    const float* b1    = (const float*)d_in[6];
    const float* W2    = (const float*)d_in[7];
    const float* b2    = (const float*)d_in[8];
    const float* W3    = (const float*)d_in[9];
    const float* b3    = (const float*)d_in[10];
    float* out = (float*)d_out;

    dim3 grid(BATCHN / BM);   // 256 workgroups of 16 batch rows
    dim3 block(256);          // 8 wave32s
    cnode_persistent_kernel<<<grid, block, LDS_BYTES, stream>>>(
        times, Y, mask, A, Bvec, W1, b1, W2, b2, W3, b3, out);
}
